// SplineFlow_71923522339354
// MI455X (gfx1250) — matrix operand order repaired and checked by reading, not compiled, not run on hardware
//
#include <hip/hip_runtime.h>
#include <cmath>

typedef __attribute__((ext_vector_type(16))) __bf16 v16bf;
typedef __attribute__((ext_vector_type(8)))  float  v8f;

#define BATCH 16
#define CCH   192
#define HALF  96
#define TT    2048
#define FF    256
#define NB    10

__device__ __forceinline__ float gelu_exact(float x) {
    return 0.5f * x * (1.0f + erff(x * 0.70710678118654752f));
}
__device__ __forceinline__ float softplusf(float x) {
    return (x > 20.0f) ? x : log1pf(expf(x));
}
__device__ __forceinline__ unsigned pack2bf(float a, float b) {
    union { __bf16 h[2]; unsigned u; } p;
    p.h[0] = (__bf16)a; p.h[1] = (__bf16)b;
    return p.u;
}

// ------------------------------------------------------------------
// C[b] = A @ B[b] + bias[m]; A row-major MxK (lda=K), B row-major KxN
// (ldb). M multiple of 128 (only used with M=256), N multiple of 128,
// K multiple of 32. Tile 128x128, 8 waves (2x4), wave = 64x32 region.
// ------------------------------------------------------------------
__global__ __launch_bounds__(256)
void gemm_bias_wmma(const float* __restrict__ A, int lda, long long a_bs,
                    const float* __restrict__ Bm, int ldb, long long b_bs,
                    const float* __restrict__ bias,
                    float* __restrict__ Cc, int ldc, long long c_bs,
                    int K)
{
    __shared__ __align__(16) __bf16 As[128 * 32];   // [m][k]
    __shared__ __align__(16) __bf16 Bs[128 * 32];   // [n][k]
    const int bz   = blockIdx.z;
    const int m0   = blockIdx.y * 128;
    const int n0   = blockIdx.x * 128;
    const int tid  = threadIdx.x;
    const int lane = tid & 31;
    const int wid  = tid >> 5;
    const int wm   = wid >> 2;   // 0..1
    const int wn   = wid & 3;    // 0..3
    const float* Ab = A  + (long long)bz * a_bs;
    const float* Bb = Bm + (long long)bz * b_bs;
    float*       Cb = Cc + (long long)bz * c_bs;

    v8f zero8;
    #pragma unroll
    for (int i = 0; i < 8; ++i) zero8[i] = 0.0f;
    v8f acc[4][2];
    #pragma unroll
    for (int mt = 0; mt < 4; ++mt)
        #pragma unroll
        for (int nt = 0; nt < 2; ++nt) acc[mt][nt] = zero8;

    // A staging map: thread -> (row = tid/2, 16 k's)
    const int a_row = tid >> 1;
    const int a_kc  = (tid & 1) << 4;
    // B staging map: thread -> (one n, 16 k's)
    const int b_n  = tid & 127;
    const int b_kh = (tid >> 7) << 4;

    const int nK = K >> 5;
    for (int kt = 0; kt < nK; ++kt) {
        { // A tile 128x32: float4 loads, packed bf16 ds_store_b128
            const float4* src =
                (const float4*)(Ab + (long long)(m0 + a_row) * lda + (kt << 5) + a_kc);
            float4 f0 = src[0], f1 = src[1], f2 = src[2], f3 = src[3];
            uint4 p0 = { pack2bf(f0.x, f0.y), pack2bf(f0.z, f0.w),
                         pack2bf(f1.x, f1.y), pack2bf(f1.z, f1.w) };
            uint4 p1 = { pack2bf(f2.x, f2.y), pack2bf(f2.z, f2.w),
                         pack2bf(f3.x, f3.y), pack2bf(f3.z, f3.w) };
            uint4* dst = (uint4*)&As[a_row * 32 + a_kc];
            dst[0] = p0; dst[1] = p1;
        }
        { // B tile 32x128 -> Bs[n][k]
            const float* src = Bb + (long long)((kt << 5) + b_kh) * ldb + n0 + b_n;
            float v[16];
            #pragma unroll
            for (int j = 0; j < 16; ++j) v[j] = src[(long long)j * ldb];
            uint4 p0 = { pack2bf(v[0], v[1]),  pack2bf(v[2], v[3]),
                         pack2bf(v[4], v[5]),  pack2bf(v[6], v[7]) };
            uint4 p1 = { pack2bf(v[8], v[9]),  pack2bf(v[10], v[11]),
                         pack2bf(v[12], v[13]), pack2bf(v[14], v[15]) };
            uint4* dst = (uint4*)&Bs[b_n * 32 + b_kh];
            dst[0] = p0; dst[1] = p1;
        }
        __syncthreads();

        v16bf afr[4], bfr[2];
        const int koff = (lane >> 4) << 3;   // 0 / 8
        #pragma unroll
        for (int mt = 0; mt < 4; ++mt) {
            int row = wm * 64 + mt * 16 + (lane & 15);
            #pragma unroll
            for (int v = 0; v < 8; ++v) {
                int k = ((v & 4) << 2) + koff + ((v & 3) << 1);
                afr[mt][2 * v]     = As[row * 32 + k];
                afr[mt][2 * v + 1] = As[row * 32 + k + 1];
            }
        }
        const int kb = (lane >> 4) << 3;     // 0 / 8
        #pragma unroll
        for (int nt = 0; nt < 2; ++nt) {
            int col = wn * 32 + nt * 16 + (lane & 15);
            #pragma unroll
            for (int v = 0; v < 8; ++v) {
                int k = ((v & 4) << 2) + kb + ((v & 3) << 1);
                bfr[nt][2 * v]     = Bs[col * 32 + k];
                bfr[nt][2 * v + 1] = Bs[col * 32 + k + 1];
            }
        }
        #pragma unroll
        for (int mt = 0; mt < 4; ++mt)
            #pragma unroll
            for (int nt = 0; nt < 2; ++nt)
                acc[mt][nt] = __builtin_amdgcn_wmma_f32_16x16x32_bf16(
                    false, afr[mt], false, bfr[nt], (short)0, acc[mt][nt],
                    false, false);
        __syncthreads();
    }

    #pragma unroll
    for (int mt = 0; mt < 4; ++mt) {
        #pragma unroll
        for (int nt = 0; nt < 2; ++nt) {
            int colg  = n0 + wn * 32 + nt * 16 + (lane & 15);
            int rbase = m0 + wm * 64 + mt * 16 + ((lane >> 4) << 3);
            #pragma unroll
            for (int r = 0; r < 8; ++r) {
                int row = rbase + r;
                Cb[(long long)row * ldc + colg] = acc[mt][nt][r] + bias[row];
            }
        }
    }
}

// ------------------------------------------------------------------
// y = gelu(channel_norm(dwconv(h*mask) + b)), one (b, 32-t tile) per block
// ------------------------------------------------------------------
__global__ __launch_bounds__(256)
void dwconv_cn_gelu(const float* __restrict__ h, const float* __restrict__ mask,
                    const float* __restrict__ w, const float* __restrict__ wb,
                    const float* __restrict__ gamma, const float* __restrict__ beta,
                    float* __restrict__ out, int dil)
{
    __shared__ float Ct[FF * 33];
    __shared__ float Sp[8 * 32], Sq[8 * 32];
    __shared__ float Mu[32], Rs[32];
    const int b  = blockIdx.y;
    const int t0 = blockIdx.x * 32;
    const int tid = threadIdx.x;
    const int ts  = tid & 31;
    const int fg  = tid >> 5;
    const int t   = t0 + ts;
    const long long base = (long long)b * FF * TT;
    const long long mb   = (long long)b * TT;

    const bool okL = (t - dil >= 0);
    const bool okR = (t + dil < TT);
    const float mm0 = okL ? mask[mb + t - dil] : 0.f;
    const float mm1 = mask[mb + t];
    const float mm2 = okR ? mask[mb + t + dil] : 0.f;

    float sm = 0.f, sq = 0.f;
    for (int i = 0; i < 32; ++i) {
        int f = fg * 32 + i;
        const float* hp = h + base + (long long)f * TT;
        float x0 = okL ? hp[t - dil] * mm0 : 0.f;
        float x1 = hp[t] * mm1;
        float x2 = okR ? hp[t + dil] * mm2 : 0.f;
        float v = w[f * 3 + 0] * x0 + w[f * 3 + 1] * x1 + w[f * 3 + 2] * x2 + wb[f];
        Ct[f * 33 + ts] = v;
        sm += v; sq += v * v;
    }
    Sp[fg * 32 + ts] = sm; Sq[fg * 32 + ts] = sq;
    __syncthreads();
    if (tid < 32) {
        float a = 0.f, c = 0.f;
        for (int g = 0; g < 8; ++g) { a += Sp[g * 32 + tid]; c += Sq[g * 32 + tid]; }
        float m   = a * (1.0f / FF);
        float var = c * (1.0f / FF) - m * m;
        Mu[tid] = m; Rs[tid] = rsqrtf(var + 1e-5f);
    }
    __syncthreads();
    float m = Mu[ts], r = Rs[ts];
    for (int i = 0; i < 32; ++i) {
        int f = fg * 32 + i;
        float v = (Ct[f * 33 + ts] - m) * r * gamma[f] + beta[f];
        out[base + (long long)f * TT + t] = gelu_exact(v);
    }
}

// ------------------------------------------------------------------
// h += gelu(channel_norm(z))
// ------------------------------------------------------------------
__global__ __launch_bounds__(256)
void cn_gelu_res(const float* __restrict__ z,
                 const float* __restrict__ gamma, const float* __restrict__ beta,
                 float* __restrict__ h)
{
    __shared__ float Ct[FF * 33];
    __shared__ float Sp[8 * 32], Sq[8 * 32];
    __shared__ float Mu[32], Rs[32];
    const int b  = blockIdx.y;
    const int t0 = blockIdx.x * 32;
    const int tid = threadIdx.x;
    const int ts  = tid & 31;
    const int fg  = tid >> 5;
    const int t   = t0 + ts;
    const long long base = (long long)b * FF * TT;

    float sm = 0.f, sq = 0.f;
    for (int i = 0; i < 32; ++i) {
        int f = fg * 32 + i;
        float v = z[base + (long long)f * TT + t];
        Ct[f * 33 + ts] = v;
        sm += v; sq += v * v;
    }
    Sp[fg * 32 + ts] = sm; Sq[fg * 32 + ts] = sq;
    __syncthreads();
    if (tid < 32) {
        float a = 0.f, c = 0.f;
        for (int g = 0; g < 8; ++g) { a += Sp[g * 32 + tid]; c += Sq[g * 32 + tid]; }
        float m   = a * (1.0f / FF);
        float var = c * (1.0f / FF) - m * m;
        Mu[tid] = m; Rs[tid] = rsqrtf(var + 1e-5f);
    }
    __syncthreads();
    float m = Mu[ts], r = Rs[ts];
    for (int i = 0; i < 32; ++i) {
        int f = fg * 32 + i;
        float v = (Ct[f * 33 + ts] - m) * r * gamma[f] + beta[f];
        h[base + (long long)f * TT + t] += gelu_exact(v);
    }
}

// ------------------------------------------------------------------
// out[:, :96] = x0 * mask
// ------------------------------------------------------------------
__global__ __launch_bounds__(256)
void copy_x0(const float* __restrict__ x, const float* __restrict__ mask,
             float* __restrict__ out, long long n)
{
    long long i = (long long)blockIdx.x * blockDim.x + threadIdx.x;
    if (i >= n) return;
    int t = (int)(i % TT);
    long long rest = i / TT;
    int c = (int)(rest % HALF);
    int b = (int)(rest / HALF);
    long long idx = ((long long)b * CCH + c) * TT + t;
    out[idx] = x[idx] * mask[(long long)b * TT + t];
}

// ------------------------------------------------------------------
// Fused proj GEMM (29x128 params from 256-dim h) + RQ spline.
// Block: one (b, channel c, 128-t tile). WMMA: M=32(29 used), N=128, K=256.
// ------------------------------------------------------------------
__global__ __launch_bounds__(256)
void proj_spline(const float* __restrict__ pw, const float* __restrict__ pb,
                 const float* __restrict__ h,  const float* __restrict__ x,
                 const float* __restrict__ mask,
                 float* __restrict__ out, float* __restrict__ ladbuf)
{
    __shared__ __align__(16) __bf16 As[32 * 32];     // [m][k]
    __shared__ __align__(16) __bf16 Bs[128 * 32];    // [n][k]
    __shared__ float  Ps[32 * 128];                  // params [row][t]
    __shared__ float  ladsh[128];
    const int b  = blockIdx.z;
    const int c  = blockIdx.y;
    const int t0 = blockIdx.x * 128;
    const int tid  = threadIdx.x;
    const int lane = tid & 31;
    const int wid  = tid >> 5;

    v8f acc0, acc1;
    #pragma unroll
    for (int i = 0; i < 8; ++i) { acc0[i] = 0.0f; acc1[i] = 0.0f; }

    // A staging map: thread -> (row = tid/8 in [0,32), 4 k's)
    const int a_row = tid >> 3;
    const int a_kc  = (tid & 7) << 2;
    // B staging map: thread -> (one n, 16 k's)
    const int b_n  = tid & 127;
    const int b_kh = (tid >> 7) << 4;
    const float mkn = mask[(long long)b * TT + t0 + b_n];

    for (int kt = 0; kt < 8; ++kt) {
        { // A: rows c*29+j (j<29, padded to 32) — uniform per-thread guard
            float4 f = {0.f, 0.f, 0.f, 0.f};
            if (a_row < 29)
                f = *(const float4*)(pw + (long long)(c * 29 + a_row) * FF +
                                     (kt << 5) + a_kc);
            uint2 p = { pack2bf(f.x, f.y), pack2bf(f.z, f.w) };
            *(uint2*)&As[a_row * 32 + a_kc] = p;
        }
        { // B: h*mask -> Bs[n][k]
            const float* src = h + ((long long)b * FF + (kt << 5) + b_kh) * TT + t0 + b_n;
            float v[16];
            #pragma unroll
            for (int j = 0; j < 16; ++j) v[j] = src[(long long)j * TT] * mkn;
            uint4 p0 = { pack2bf(v[0], v[1]),  pack2bf(v[2], v[3]),
                         pack2bf(v[4], v[5]),  pack2bf(v[6], v[7]) };
            uint4 p1 = { pack2bf(v[8], v[9]),  pack2bf(v[10], v[11]),
                         pack2bf(v[12], v[13]), pack2bf(v[14], v[15]) };
            uint4* dst = (uint4*)&Bs[b_n * 32 + b_kh];
            dst[0] = p0; dst[1] = p1;
        }
        __syncthreads();

        v16bf afr0, afr1, bfr;
        const int koff = (lane >> 4) << 3;
        const int r0   = lane & 15;
        #pragma unroll
        for (int v = 0; v < 8; ++v) {
            int k = ((v & 4) << 2) + koff + ((v & 3) << 1);
            afr0[2 * v]     = As[r0 * 32 + k];
            afr0[2 * v + 1] = As[r0 * 32 + k + 1];
            afr1[2 * v]     = As[(16 + r0) * 32 + k];
            afr1[2 * v + 1] = As[(16 + r0) * 32 + k + 1];
        }
        const int col = wid * 16 + (lane & 15);
        const int kb  = (lane >> 4) << 3;
        #pragma unroll
        for (int v = 0; v < 8; ++v) {
            int k = ((v & 4) << 2) + kb + ((v & 3) << 1);
            bfr[2 * v]     = Bs[col * 32 + k];
            bfr[2 * v + 1] = Bs[col * 32 + k + 1];
        }
        acc0 = __builtin_amdgcn_wmma_f32_16x16x32_bf16(
            false, afr0, false, bfr, (short)0, acc0, false, false);
        acc1 = __builtin_amdgcn_wmma_f32_16x16x32_bf16(
            false, afr1, false, bfr, (short)0, acc1, false, false);
        __syncthreads();
    }

    { // accumulators -> Ps with bias
        int col = wid * 16 + (lane & 15);
        int rb  = (lane >> 4) << 3;
        #pragma unroll
        for (int r = 0; r < 8; ++r) {
            int row0 = rb + r;          // < 16
            Ps[row0 * 128 + col] = acc0[r] + pb[c * 29 + row0];
            int row1 = 16 + rb + r;     // 16..31 (29..31 unused)
            float bz = (row1 < 29) ? pb[c * 29 + row1] : 0.f;
            Ps[row1 * 128 + col] = acc1[r] + bz;
        }
    }
    __syncthreads();

    float ladv = 0.f;
    if (tid < 128) {
        const int t = t0 + tid;
        float uwv[NB], uhv[NB], udv[NB - 1];
        #pragma unroll
        for (int j = 0; j < NB; ++j)     uwv[j] = Ps[j * 128 + tid] * 0.0625f;
        #pragma unroll
        for (int j = 0; j < NB; ++j)     uhv[j] = Ps[(NB + j) * 128 + tid] * 0.0625f;
        #pragma unroll
        for (int j = 0; j < NB - 1; ++j) udv[j] = Ps[(2 * NB + j) * 128 + tid];

        // widths -> cumwidths
        float mxw = uwv[0];
        #pragma unroll
        for (int j = 1; j < NB; ++j) mxw = fmaxf(mxw, uwv[j]);
        float ew[NB], sw = 0.f;
        #pragma unroll
        for (int j = 0; j < NB; ++j) { ew[j] = expf(uwv[j] - mxw); sw += ew[j]; }
        float invw = 1.0f / sw;
        float cw[NB + 1]; cw[0] = 0.f;
        #pragma unroll
        for (int j = 0; j < NB; ++j) cw[j + 1] = cw[j] + 0.001f + 0.99f * ew[j] * invw;
        #pragma unroll
        for (int j = 0; j <= NB; ++j) cw[j] = 10.f * cw[j] - 5.f;
        cw[0] = -5.f; cw[NB] = 5.f;

        // heights -> cumheights
        float mxh = uhv[0];
        #pragma unroll
        for (int j = 1; j < NB; ++j) mxh = fmaxf(mxh, uhv[j]);
        float eh[NB], sh = 0.f;
        #pragma unroll
        for (int j = 0; j < NB; ++j) { eh[j] = expf(uhv[j] - mxh); sh += eh[j]; }
        float invh = 1.0f / sh;
        float ch[NB + 1]; ch[0] = 0.f;
        #pragma unroll
        for (int j = 0; j < NB; ++j) ch[j + 1] = ch[j] + 0.001f + 0.99f * eh[j] * invh;
        #pragma unroll
        for (int j = 0; j <= NB; ++j) ch[j] = 10.f * ch[j] - 5.f;
        ch[0] = -5.f; ch[NB] = 5.f;

        // derivatives (padded with log(expm1(1-MIN_D)))
        const float dconst = logf(expm1f(0.999f));
        float derivs[NB + 1];
        derivs[0] = 0.001f + softplusf(dconst);
        derivs[NB] = derivs[0];
        #pragma unroll
        for (int j = 0; j < NB - 1; ++j) derivs[j + 1] = 0.001f + softplusf(udv[j]);

        const float xv = x[((long long)b * CCH + HALF + c) * TT + t];
        const float mk = mask[(long long)b * TT + t];
        const bool inside = (xv >= -5.f) && (xv <= 5.f);
        float xc = fminf(fmaxf(xv, -5.f), 5.f);
        int idx = 0;
        #pragma unroll
        for (int j = 1; j < NB; ++j) if (xc >= cw[j]) idx = j;

        float icw = cw[idx], w_ = cw[idx + 1] - cw[idx];
        float ich = ch[idx], h_ = ch[idx + 1] - ch[idx];
        float delta = h_ / w_;
        float d0 = derivs[idx], d1 = derivs[idx + 1];
        float th  = (xc - icw) / w_;
        float t1m = th * (1.f - th);
        float denom = delta + (d0 + d1 - 2.f * delta) * t1m;
        float num   = h_ * (delta * th * th + d0 * t1m);
        float outv  = ich + num / denom;
        float dnum  = delta * delta *
                      (d1 * th * th + 2.f * delta * t1m + d0 * (1.f - th) * (1.f - th));
        float lad   = logf(dnum) - 2.f * logf(denom);

        out[((long long)b * CCH + HALF + c) * TT + t] = (inside ? outv : xv) * mk;
        ladv = (inside ? lad : 0.f) * mk;
    }
    if (tid < 128) ladsh[tid] = ladv;
    __syncthreads();
    for (int s = 64; s > 0; s >>= 1) {
        if (tid < s) ladsh[tid] += ladsh[tid + s];
        __syncthreads();
    }
    if (tid == 0)
        ladbuf[((long long)b * HALF + c) * gridDim.x + blockIdx.x] = ladsh[0];
}

// ------------------------------------------------------------------
// logdet[b] = sum over 96*16 per-block partials (deterministic)
// ------------------------------------------------------------------
__global__ __launch_bounds__(256)
void reduce_logdet(const float* __restrict__ ladbuf, float* __restrict__ logdet, int n)
{
    __shared__ float s[256];
    const int b = blockIdx.x;
    float a = 0.f;
    for (int i = threadIdx.x; i < n; i += 256) a += ladbuf[(long long)b * n + i];
    s[threadIdx.x] = a;
    __syncthreads();
    for (int st = 128; st > 0; st >>= 1) {
        if (threadIdx.x < st) s[threadIdx.x] += s[threadIdx.x + st];
        __syncthreads();
    }
    if (threadIdx.x == 0) logdet[b] = s[0];
}

extern "C" void kernel_launch(void* const* d_in, const int* in_sizes, int n_in,
                              void* d_out, int out_size, void* d_ws, size_t ws_size,
                              hipStream_t stream) {
    const float* x      = (const float*)d_in[0];
    const float* xmask  = (const float*)d_in[1];
    const float* pre_w  = (const float*)d_in[2];
    const float* pre_b  = (const float*)d_in[3];
    const float* dw_w   = (const float*)d_in[4];
    const float* dw_b   = (const float*)d_in[5];
    const float* pw_w   = (const float*)d_in[6];
    const float* pw_b   = (const float*)d_in[7];
    const float* g1     = (const float*)d_in[8];
    const float* b1     = (const float*)d_in[9];
    const float* g2     = (const float*)d_in[10];
    const float* b2     = (const float*)d_in[11];
    const float* proj_w = (const float*)d_in[12];
    const float* proj_b = (const float*)d_in[13];
    float* out = (float*)d_out;

    const long long FT = (long long)FF * TT;   // 524288
    float* h = (float*)d_ws;
    float* y = h + (long long)BATCH * FT;
    float* z = y + (long long)BATCH * FT;
    float* ladbuf = z + (long long)BATCH * FT;

    // h = pre_w @ x0 + pre_b
    gemm_bias_wmma<<<dim3(TT / 128, FF / 128, BATCH), 256, 0, stream>>>(
        pre_w, HALF, 0LL, x, TT, (long long)CCH * TT, pre_b,
        h, TT, FT, HALF);

    int dil = 1;
    for (int i = 0; i < 3; ++i) {
        dwconv_cn_gelu<<<dim3(TT / 32, BATCH), 256, 0, stream>>>(
            h, xmask, dw_w + (long long)i * FF * 3, dw_b + (long long)i * FF,
            g1 + (long long)i * FF, b1 + (long long)i * FF, y, dil);
        gemm_bias_wmma<<<dim3(TT / 128, FF / 128, BATCH), 256, 0, stream>>>(
            pw_w + (long long)i * FF * FF, FF, 0LL, y, TT, FT,
            pw_b + (long long)i * FF, z, TT, FT, FF);
        cn_gelu_res<<<dim3(TT / 32, BATCH), 256, 0, stream>>>(
            z, g2 + (long long)i * FF, b2 + (long long)i * FF, h);
        dil *= 3;
    }

    long long n0 = (long long)BATCH * HALF * TT;
    copy_x0<<<(unsigned)((n0 + 255) / 256), 256, 0, stream>>>(x, xmask, out, n0);

    proj_spline<<<dim3(TT / 128, HALF, BATCH), 256, 0, stream>>>(
        proj_w, proj_b, h, x, xmask, out, ladbuf);

    reduce_logdet<<<BATCH, 256, 0, stream>>>(
        ladbuf, out + (long long)BATCH * CCH * TT, HALF * (TT / 128));
}
